// RCD_69114613730662
// MI455X (gfx1250) — matrix operand order repaired and checked by reading, not compiled, not run on hardware
//
#include <hip/hip_runtime.h>
#include <hip/hip_bf16.h>
#include <math.h>

// ---------------- problem constants ----------------
#define SS     100000
#define EE     5000
#define KK     68
#define BB     4096
#define NEDGE  1000000
#define HH     512
#define SENS_MEAN_F 0.00014507418272432547f
#define BN_EPS 1e-5f

typedef __attribute__((ext_vector_type(16))) _Float16 v16h;
typedef __attribute__((ext_vector_type(8)))  _Float16 v8h;
typedef __attribute__((ext_vector_type(8)))  float    v8f;

__device__ __forceinline__ float sigf(float x) { return 1.0f / (1.0f + expf(-x)); }

// native f32 atomic add (global_atomic_add_f32) — avoid CAS-loop fallback
__device__ __forceinline__ void atomAddF(float* p, float v) { unsafeAtomicAdd(p, v); }

// ---------------- generic utility kernels ----------------
__global__ void k_fill(float* p, float v, long n) {
    long i = (long)blockIdx.x * 256 + threadIdx.x;
    if (i < n) p[i] = v;
}
__global__ void k_copy(float* dst, const float* src, long n) {
    long i = (long)blockIdx.x * 256 + threadIdx.x;
    if (i < n) dst[i] = src[i];
}
__global__ void k_accum(float* sum, const float* x, long n) {
    long i = (long)blockIdx.x * 256 + threadIdx.x;
    if (i < n) sum[i] += x[i];
}

// ---------------- graph-conv kernels ----------------
__global__ void k_degree(const int* __restrict__ idx, float* deg, int n) {
    int i = blockIdx.x * 256 + threadIdx.x;
    if (i < n) atomAddF(&deg[idx[i]], 1.0f);
}
__global__ void k_recip1p(float* d, int n) {
    int i = blockIdx.x * 256 + threadIdx.x;
    if (i < n) d[i] = 1.0f / (d[i] + 1.0f);
}
__global__ void k_vals(const int* __restrict__ es, const int* __restrict__ ee,
                       const float* __restrict__ du, const float* __restrict__ di,
                       float* vals, int n) {
    int i = blockIdx.x * 256 + threadIdx.x;
    if (i < n) vals[i] = sqrtf(du[es[i]] * di[ee[i]]);
}
// dst[r*K+k] = src[r*K+k] * d[r]   (init buffer with prev*deg before atomic SpMM adds)
__global__ void k_scale_rows(float* dst, const float* __restrict__ src,
                             const float* __restrict__ d, long n) {
    long i = (long)blockIdx.x * 256 + threadIdx.x;
    if (i < n) dst[i] = src[i] * d[i / KK];
}
// edge-parallel scatter, one thread per (edge, 4-wide k chunk): 16B gathers,
// native f32 atomics into an L2-resident destination table.
__global__ void k_spmm(float* dst, const float* __restrict__ src,
                       const int* __restrict__ idx_dst, const int* __restrict__ idx_src,
                       const float* __restrict__ vals, int ne) {
    const int KC = KK / 4;  // 17 chunks
    long gid = (long)blockIdx.x * 256 + threadIdx.x;
    if (gid >= (long)ne * KC) return;
    int e  = (int)(gid / KC);
    int k4 = (int)(gid - (long)e * KC) * 4;
    float v = vals[e];
    const float* s = src + (long)idx_src[e] * KK + k4;
    float*       d = dst + (long)idx_dst[e] * KK + k4;
    float4 x = *(const float4*)s;
    atomAddF(d + 0, v * x.x);
    atomAddF(d + 1, v * x.y);
    atomAddF(d + 2, v * x.z);
    atomAddF(d + 3, v * x.w);
}

// ---------------- staging kernels (f32 -> padded f16) ----------------
// weights: out[Np x Kp], zero pad; absmode => |W| (pos_lin identity 2*relu(-W)+W == |W|)
__global__ void k_stage_w(const float* __restrict__ W, _Float16* out,
                          int N, int Kd, int Kp, int absmode, long n) {
    long i = (long)blockIdx.x * 256 + threadIdx.x;
    if (i >= n) return;
    int r = (int)(i / Kp), c = (int)(i - (long)r * Kp);
    float v = (r < N && c < Kd) ? W[(long)r * Kd + c] : 0.0f;
    if (absmode) v = fabsf(v);
    out[i] = (_Float16)v;
}
// B x Kp activation from a length-B vector in column 0
__global__ void k_vec_col_f16(const float* __restrict__ v, _Float16* out, int kp, long n) {
    long i = (long)blockIdx.x * 256 + threadIdx.x;
    if (i >= n) return;
    int b = (int)(i / kp), k = (int)(i - (long)b * kp);
    out[i] = (_Float16)((k == 0) ? v[b] : 0.0f);
}
// Xcat[b, 0:68]=stu_emb[stu_id[b]], [68:136]=sens_feat[b], pad->0, Kp=160
__global__ void k_concat(const float* __restrict__ stu_emb, const int* __restrict__ stu_id,
                         const float* __restrict__ sens_feat, _Float16* out, long n) {
    long i = (long)blockIdx.x * 256 + threadIdx.x;
    if (i >= n) return;
    int b = (int)(i / 160), k = (int)(i - (long)b * 160);
    float v = 0.0f;
    if (k < KK)            v = stu_emb[(long)stu_id[b] * KK + k];
    else if (k < 2 * KK)   v = sens_feat[(long)b * KK + (k - KK)];
    out[i] = (_Float16)v;
}

// ---------------- WMMA GEMM:  C[M x Np] = A[M x Kp] @ W[Np x Kp]^T + bias ----------------
// one 16x16 C tile per wave; v_wmma_f32_16x16x32_f16, f32 accumulate.
// A fragment (16x32): lane L holds row m=L&15; halves j = K [(L>>4)*16 + j], contiguous.
// B fragment (32x16): lane L holds column n=L&15; halves j = K [(L>>4)*16 + j], contiguous.
__global__ __launch_bounds__(256) void k_gemm(const _Float16* __restrict__ A,
                                              const _Float16* __restrict__ W,
                                              const float* __restrict__ bias,
                                              float* __restrict__ C,
                                              int M, int Nreal, int Np, int Kp) {
    int wave = threadIdx.x >> 5;
    int lane = threadIdx.x & 31;
    int mt_cnt = M >> 4;
    int tile = blockIdx.x * 8 + wave;
    int nt = tile / mt_cnt;
    int mt = tile - nt * mt_cnt;
    if (nt >= (Np >> 4)) return;               // wave-uniform: EXEC stays all-ones for WMMA
    int l15 = lane & 15;
    int kb  = lane >> 4;
    const _Float16* rowA = A + (long)(mt * 16 + l15) * Kp;
    const _Float16* rowW = W + (long)(nt * 16 + l15) * Kp;
    v8f acc = {};
    for (int kk = 0; kk < Kp; kk += 32) {
        v16h a = *(const v16h*)(rowA + kk + kb * 16);
        v16h b = *(const v16h*)(rowW + kk + kb * 16);
        acc = __builtin_amdgcn_wmma_f32_16x16x32_f16(false, a, false, b,
                                                     (short)0, acc, false, false);
    }
    int col = nt * 16 + l15;
    float bb = (col < Nreal) ? bias[col] : 0.0f;
#pragma unroll
    for (int r = 0; r < 8; ++r) {              // C layout: VGPR r -> M = r + 8*kb, N = lane&15
        int row = mt * 16 + r + kb * 8;
        C[(long)row * Np + col] = acc[r] + bb;
    }
}

// ---------------- batchnorm ----------------
__global__ void k_bn_stats(const float* __restrict__ X, int rows, int ld,
                           float* mean, float* rstd) {
    __shared__ float s1[256], s2[256];
    int n = blockIdx.x, t = threadIdx.x;
    float a = 0.f, b = 0.f;
    for (int r = t; r < rows; r += 256) {
        float v = X[(long)r * ld + n];
        a += v; b += v * v;
    }
    s1[t] = a; s2[t] = b; __syncthreads();
    for (int o = 128; o > 0; o >>= 1) {
        if (t < o) { s1[t] += s1[t + o]; s2[t] += s2[t + o]; }
        __syncthreads();
    }
    if (t == 0) {
        float m = s1[0] / rows;
        float v = s2[0] / rows - m * m;
        if (v < 0.f) v = 0.f;
        mean[n] = m;
        rstd[n] = rsqrtf(v + BN_EPS);
    }
}
// y = g*(x-m)*rstd + s ; optional relu; optional f32 out (ldo) and padded f16 out (kp)
__global__ void k_bn_apply(const float* __restrict__ X, int cols, int ld,
                           const float* __restrict__ mean, const float* __restrict__ rstd,
                           const float* __restrict__ g, const float* __restrict__ s,
                           float* o32, int ldo, _Float16* o16, int kp, int relu, long n) {
    long i = (long)blockIdx.x * 256 + threadIdx.x;
    if (i >= n) return;
    int b = (int)(i / kp), k = (int)(i - (long)b * kp);
    if (k < cols) {
        float y = g[k] * (X[(long)b * ld + k] - mean[k]) * rstd[k] + s[k];
        if (relu && y < 0.f) y = 0.f;
        if (o32) o32[(long)b * ldo + k] = y;
        if (o16) o16[(long)b * kp + k] = (_Float16)y;
    } else if (o16) {
        o16[(long)b * kp + k] = (_Float16)0.0f;
    }
}

// sigmoid -> padded f16 (pred-head intermediates, no BN)
__global__ void k_sig_f16(const float* __restrict__ X, int cols, int ld,
                          _Float16* o16, int kp, long n) {
    long i = (long)blockIdx.x * 256 + threadIdx.x;
    if (i >= n) return;
    int b = (int)(i / kp), k = (int)(i - (long)b * kp);
    o16[i] = (_Float16)((k < cols) ? sigf(X[(long)b * ld + k]) : 0.0f);
}

// fused stat/debias/multi-hot -> x in padded f16 (B x 96)
// con_Uf = sigmoid(s3_combine) exactly (BN of identical-row batch == shift vector)
__global__ void k_stat_x(const float* __restrict__ UfF, const float* __restrict__ UdF,
                         const float* __restrict__ alphaP, const float* __restrict__ betaP,
                         const int* __restrict__ stu_id, const float* __restrict__ s3c,
                         const float* __restrict__ exer_emb, const int* __restrict__ input_exer,
                         const float* __restrict__ e_diffP, const int* __restrict__ kp_idx,
                         _Float16* xout, long n) {
    long i = (long)blockIdx.x * 256 + threadIdx.x;
    if (i >= n) return;
    int b = (int)(i / 96), k = (int)(i - (long)b * 96);
    if (k >= KK) { xout[i] = (_Float16)0.0f; return; }
    int sid = stu_id[b], eid = input_exer[b];
    float al = sigf(alphaP[sid]);
    float be = sigf(betaP[sid]);
    float Uf = sigf(UfF[(long)b * KK + k]);
    float Ud = sigf(UdF[(long)b * KK + k]);
    float stat = sigf((1.f - al) * Uf + al * Ud);
    float conUf = sigf(s3c[k]);
    float constat = sigf((1.f - al) * conUf + al * Ud);
    float debias = sigf(stat - be * constat);
    float kd = sigf(exer_emb[(long)eid * KK + k]);
    float ed = sigf(e_diffP[eid]);
    const int* kp = kp_idx + (long)b * 4;
    float mh = (kp[0] == k || kp[1] == k || kp[2] == k || kp[3] == k) ? 1.0f : 0.0f;
    xout[i] = (_Float16)(ed * (debias - kd) * mh);
}

__global__ void k_final_out(const float* __restrict__ C, float* out, int n) {
    int i = blockIdx.x * 256 + threadIdx.x;
    if (i < n) out[i] = sigf(C[(long)i * 16]);
}

// reverse_loss = mean((UfRev-SENS_MEAN)^2) + mean((UdRev-sens)^2)
__global__ void k_rev_loss(const float* __restrict__ ufr, const float* __restrict__ udr,
                           const float* __restrict__ sens, float* out, int n) {
    __shared__ float sh[1024];
    int t = threadIdx.x;
    float a = 0.f;
    for (int i = t; i < n; i += 1024) {
        float d1 = ufr[i] - SENS_MEAN_F;
        float d2 = udr[i] - sens[i];
        a += d1 * d1 + d2 * d2;
    }
    sh[t] = a; __syncthreads();
    for (int o = 512; o > 0; o >>= 1) {
        if (t < o) sh[t] += sh[t + o];
        __syncthreads();
    }
    if (t == 0) out[0] = sh[0] / (float)n;
}

// cls_loss partial: acc += scale * sum_b [softplus(sig(z)) - sig(z)*y]
__global__ void k_cls_loss(const float* __restrict__ lin, int ld,
                           const int* __restrict__ labels, float* acc, int n, float scale) {
    __shared__ float sh[512];
    int t = threadIdx.x;
    float a = 0.f;
    for (int i = t; i < n; i += 512) {
        float z = sigf(lin[(long)i * ld]);
        float y = (float)labels[i];
        a += log1pf(expf(z)) - z * y;
    }
    sh[t] = a; __syncthreads();
    for (int o = 256; o > 0; o >>= 1) {
        if (t < o) sh[t] += sh[t + o];
        __syncthreads();
    }
    if (t == 0) atomAddF(acc, sh[0] * scale);
}

// ---------------- host helpers ----------------
static inline int cdiv(long a, int b) { return (int)((a + b - 1) / b); }

static void run_gemm(hipStream_t st, const _Float16* A, const float* W, const float* bias,
                     float* C, int M, int Nreal, int Np, int Kreal, int Kp, int absmode,
                     _Float16* wbuf) {
    long wn = (long)Np * Kp;
    k_stage_w<<<cdiv(wn, 256), 256, 0, st>>>(W, wbuf, Nreal, Kreal, Kp, absmode, wn);
    int tiles = (M / 16) * (Np / 16);
    k_gemm<<<cdiv(tiles, 8), 256, 0, st>>>(A, wbuf, bias, C, M, Nreal, Np, Kp);
}

static void run_bn(hipStream_t st, const float* C, int M, int cols, int ld,
                   const float* g, const float* s, float* mean, float* rstd,
                   float* o32, int ldo, _Float16* o16, int kp, int relu) {
    k_bn_stats<<<cols, 256, 0, st>>>(C, M, ld, mean, rstd);
    int kpp = o16 ? kp : cols;
    long n = (long)M * kpp;
    k_bn_apply<<<cdiv(n, 256), 256, 0, st>>>(C, cols, ld, mean, rstd, g, s,
                                             o32, ldo, o16, kpp, relu, n);
}

extern "C" void kernel_launch(void* const* d_in, const int* in_sizes, int n_in,
                              void* d_out, int out_size, void* d_ws, size_t ws_size,
                              hipStream_t stream) {
    // ---- input mapping: recursive insertion order of setup_inputs() dict ----
    const int*   stu_id      = (const int*)d_in[0];
    const int*   input_exer  = (const int*)d_in[1];
    const int*   input_kp    = (const int*)d_in[2];
    const float* sensitive   = (const float*)d_in[3];
    /* labels d_in[4] unused by outputs */
    const int*   cls_labels  = (const int*)d_in[5];
    const int*   edge_stu    = (const int*)d_in[6];
    const int*   edge_exer   = (const int*)d_in[7];
    const float* student_emb = (const float*)d_in[8];
    const float* kdiff_emb   = (const float*)d_in[9];
    const float* e_diffP     = (const float*)d_in[10];
    const float* alphaP      = (const float*)d_in[11];
    const float* betaP       = (const float*)d_in[12];
    // blocks: W1,b1,g1,s1,W2,b2,g2,s2[,W3,b3,g3,s3]
    const int iComb = 13, iSens = 25, iSenD = 37, iRev = 49, iCls = 57;
    const int iP1 = 97, iP2 = 99, iP3 = 101;
#define PF(i) ((const float*)d_in[(i)])

    float* out = (float*)d_out;

    // ---- workspace bump allocator ----
    char* wp = (char*)d_ws;
    auto alloc = [&](size_t bytes) -> void* {
        void* r = (void*)wp;
        wp += (bytes + 255) & ~(size_t)255;
        return r;
    };
    const long SK = (long)SS * KK, EK = (long)EE * KK;
    const long NEK4 = (long)NEDGE * (KK / 4);
    float* du    = (float*)alloc(sizeof(float) * SS);
    float* dI    = (float*)alloc(sizeof(float) * EE);
    float* vals  = (float*)alloc(sizeof(float) * NEDGE);
    float* sum_u = (float*)alloc(sizeof(float) * SK);
    float* u_a   = (float*)alloc(sizeof(float) * SK);
    float* u_b   = (float*)alloc(sizeof(float) * SK);
    float* sum_i = (float*)alloc(sizeof(float) * EK);
    float* i_a   = (float*)alloc(sizeof(float) * EK);
    float* i_b   = (float*)alloc(sizeof(float) * EK);
    _Float16* wbuf = (_Float16*)alloc(2ull * 512 * 512);
    _Float16* fA   = (_Float16*)alloc(2ull * BB * 512);
    _Float16* fB   = (_Float16*)alloc(2ull * BB * 512);
    _Float16* Uf16 = (_Float16*)alloc(2ull * BB * 96);
    _Float16* Ud16 = (_Float16*)alloc(2ull * BB * 96);
    _Float16* sf16 = (_Float16*)alloc(2ull * BB * 96);
    float* C0 = (float*)alloc(4ull * BB * 512);
    float* C1 = (float*)alloc(4ull * BB * 512);
    float* sens_feat = (float*)alloc(4ull * BB * KK);
    float* UfF   = (float*)alloc(4ull * BB * KK);
    float* UdF   = (float*)alloc(4ull * BB * KK);
    float* UfRev = (float*)alloc(4ull * BB);
    float* UdRev = (float*)alloc(4ull * BB);
    float* meanb = (float*)alloc(4ull * 512);
    float* rstdb = (float*)alloc(4ull * 512);

    // ================= graph convolution =================
    k_fill<<<cdiv(SS, 256), 256, 0, stream>>>(du, 0.f, SS);
    k_fill<<<cdiv(EE, 256), 256, 0, stream>>>(dI, 0.f, EE);
    k_degree<<<cdiv(NEDGE, 256), 256, 0, stream>>>(edge_stu, du, NEDGE);
    k_degree<<<cdiv(NEDGE, 256), 256, 0, stream>>>(edge_exer, dI, NEDGE);
    k_recip1p<<<cdiv(SS, 256), 256, 0, stream>>>(du, SS);
    k_recip1p<<<cdiv(EE, 256), 256, 0, stream>>>(dI, EE);
    k_vals<<<cdiv(NEDGE, 256), 256, 0, stream>>>(edge_stu, edge_exer, du, dI, vals, NEDGE);

    auto spmm_ui = [&](float* dst, const float* src) {
        k_spmm<<<cdiv(NEK4, 256), 256, 0, stream>>>(dst, src, edge_stu, edge_exer, vals, NEDGE);
    };
    auto spmm_iu = [&](float* dst, const float* src) {
        k_spmm<<<cdiv(NEK4, 256), 256, 0, stream>>>(dst, src, edge_exer, edge_stu, vals, NEDGE);
    };
    auto scaleU = [&](float* dst, const float* src) {
        k_scale_rows<<<cdiv(SK, 256), 256, 0, stream>>>(dst, src, du, SK);
    };
    auto scaleI = [&](float* dst, const float* src) {
        k_scale_rows<<<cdiv(EK, 256), 256, 0, stream>>>(dst, src, dI, EK);
    };
    auto accU = [&](float* x) { k_accum<<<cdiv(SK, 256), 256, 0, stream>>>(sum_u, x, SK); };
    auto accI = [&](float* x) { k_accum<<<cdiv(EK, 256), 256, 0, stream>>>(sum_i, x, EK); };

    // u1 / i1
    scaleU(u_a, student_emb);  spmm_ui(u_a, kdiff_emb);
    k_copy<<<cdiv(SK, 256), 256, 0, stream>>>(sum_u, student_emb, SK);  accU(u_a);
    scaleI(i_a, kdiff_emb);    spmm_iu(i_a, student_emb);
    k_copy<<<cdiv(EK, 256), 256, 0, stream>>>(sum_i, kdiff_emb, EK);    accI(i_a);
    // u2 = spmm_ui(i1) + u1*du ; i2 = spmm_iu(u2) + i1*di
    scaleU(u_b, u_a); spmm_ui(u_b, i_a); accU(u_b);
    scaleI(i_b, i_a); spmm_iu(i_b, u_b); accI(i_b);
    // u3 = spmm_ui(i2) + u2*du -> u_a ; i3 = spmm_iu(u2!) + i2*di -> i_a
    scaleU(u_a, u_b); spmm_ui(u_a, i_b); accU(u_a);
    scaleI(i_a, i_b); spmm_iu(i_a, u_b); accI(i_a);
    // u4 = spmm_ui(i3) + u3*du -> u_b ; i4 = spmm_iu(u3) + i3*di -> i_b
    scaleU(u_b, u_a); spmm_ui(u_b, i_a); accU(u_b);
    scaleI(i_b, i_a); spmm_iu(i_b, u_a); accI(i_b);
    // sum_u = stu_emb, sum_i = exer_emb

    // ================= batch MLP pipeline (WMMA) =================
    // ---- sens MLP3: 1 -> 512 -> 256 -> 68 ----
    {
        long n = (long)BB * 32;
        k_vec_col_f16<<<cdiv(n, 256), 256, 0, stream>>>(sensitive, fA, 32, n);
    }
    run_gemm(stream, fA, PF(iSens + 0), PF(iSens + 1), C0, BB, 512, 512, 1, 32, 0, wbuf);
    run_bn(stream, C0, BB, 512, 512, PF(iSens + 2), PF(iSens + 3), meanb, rstdb,
           nullptr, 0, fB, 512, 1);
    run_gemm(stream, fB, PF(iSens + 4), PF(iSens + 5), C1, BB, 256, 256, 512, 512, 0, wbuf);
    run_bn(stream, C1, BB, 256, 256, PF(iSens + 6), PF(iSens + 7), meanb, rstdb,
           nullptr, 0, fA, 256, 1);
    run_gemm(stream, fA, PF(iSens + 8), PF(iSens + 9), C0, BB, 68, 80, 256, 256, 0, wbuf);
    run_bn(stream, C0, BB, 68, 80, PF(iSens + 10), PF(iSens + 11), meanb, rstdb,
           sens_feat, 68, sf16, 96, 0);

    // ---- combine MLP3: concat(stu_emb[stu_id], sens_feat) 136 -> 512 -> 256 -> 68 ----
    {
        long n = (long)BB * 160;
        k_concat<<<cdiv(n, 256), 256, 0, stream>>>(sum_u, stu_id, sens_feat, fA, n);
    }
    run_gemm(stream, fA, PF(iComb + 0), PF(iComb + 1), C0, BB, 512, 512, 136, 160, 0, wbuf);
    run_bn(stream, C0, BB, 512, 512, PF(iComb + 2), PF(iComb + 3), meanb, rstdb,
           nullptr, 0, fB, 512, 1);
    run_gemm(stream, fB, PF(iComb + 4), PF(iComb + 5), C1, BB, 256, 256, 512, 512, 0, wbuf);
    run_bn(stream, C1, BB, 256, 256, PF(iComb + 6), PF(iComb + 7), meanb, rstdb,
           nullptr, 0, fA, 256, 1);
    run_gemm(stream, fA, PF(iComb + 8), PF(iComb + 9), C0, BB, 68, 80, 256, 256, 0, wbuf);
    run_bn(stream, C0, BB, 68, 80, PF(iComb + 10), PF(iComb + 11), meanb, rstdb,
           UfF, 68, Uf16, 96, 0);

    // ---- sens_dense MLP3: 68 -> 512 -> 256 -> 68 ----
    run_gemm(stream, sf16, PF(iSenD + 0), PF(iSenD + 1), C0, BB, 512, 512, 68, 96, 0, wbuf);
    run_bn(stream, C0, BB, 512, 512, PF(iSenD + 2), PF(iSenD + 3), meanb, rstdb,
           nullptr, 0, fB, 512, 1);
    run_gemm(stream, fB, PF(iSenD + 4), PF(iSenD + 5), C1, BB, 256, 256, 512, 512, 0, wbuf);
    run_bn(stream, C1, BB, 256, 256, PF(iSenD + 6), PF(iSenD + 7), meanb, rstdb,
           nullptr, 0, fA, 256, 1);
    run_gemm(stream, fA, PF(iSenD + 8), PF(iSenD + 9), C0, BB, 68, 80, 256, 256, 0, wbuf);
    run_bn(stream, C0, BB, 68, 80, PF(iSenD + 10), PF(iSenD + 11), meanb, rstdb,
           UdF, 68, Ud16, 96, 0);

    // ---- sens_rev on Uf_features and Ud_features: 68 -> 512 -> 1 ----
    run_gemm(stream, Uf16, PF(iRev + 0), PF(iRev + 1), C0, BB, 512, 512, 68, 96, 0, wbuf);
    run_bn(stream, C0, BB, 512, 512, PF(iRev + 2), PF(iRev + 3), meanb, rstdb,
           nullptr, 0, fB, 512, 1);
    run_gemm(stream, fB, PF(iRev + 4), PF(iRev + 5), C1, BB, 1, 16, 512, 512, 0, wbuf);
    run_bn(stream, C1, BB, 1, 16, PF(iRev + 6), PF(iRev + 7), meanb, rstdb,
           UfRev, 1, nullptr, 0, 0);

    run_gemm(stream, Ud16, PF(iRev + 0), PF(iRev + 1), C0, BB, 512, 512, 68, 96, 0, wbuf);
    run_bn(stream, C0, BB, 512, 512, PF(iRev + 2), PF(iRev + 3), meanb, rstdb,
           nullptr, 0, fB, 512, 1);
    run_gemm(stream, fB, PF(iRev + 4), PF(iRev + 5), C1, BB, 1, 16, 512, 512, 0, wbuf);
    run_bn(stream, C1, BB, 1, 16, PF(iRev + 6), PF(iRev + 7), meanb, rstdb,
           UdRev, 1, nullptr, 0, 0);

    // ---- 5 classifiers on Uf_features (no BN on layer 2) ----
    k_fill<<<1, 256, 0, stream>>>(out + BB + 1, 0.f, 1);
    for (int c = 0; c < 5; ++c) {
        int base = iCls + 8 * c;
        run_gemm(stream, Uf16, PF(base + 0), PF(base + 1), C0, BB, 512, 512, 68, 96, 0, wbuf);
        run_bn(stream, C0, BB, 512, 512, PF(base + 2), PF(base + 3), meanb, rstdb,
               nullptr, 0, fB, 512, 1);
        run_gemm(stream, fB, PF(base + 4), PF(base + 5), C1, BB, 1, 16, 512, 512, 0, wbuf);
        k_cls_loss<<<1, 512, 0, stream>>>(C1, 16, cls_labels + (long)c * BB,
                                          out + BB + 1, BB, 1.0f / (5.0f * BB));
    }

    // ---- fused stat/debias/multi-hot -> prediction head (pos_lin == |W|) ----
    {
        long n = (long)BB * 96;
        k_stat_x<<<cdiv(n, 256), 256, 0, stream>>>(UfF, UdF, alphaP, betaP, stu_id,
                                                   PF(iComb + 11) /* s3_combine */,
                                                   sum_i, input_exer, e_diffP, input_kp, fA, n);
    }
    run_gemm(stream, fA, PF(iP1), PF(iP1 + 1), C0, BB, 512, 512, 68, 96, 1, wbuf);
    {
        long n = (long)BB * 512;
        k_sig_f16<<<cdiv(n, 256), 256, 0, stream>>>(C0, 512, 512, fB, 512, n);
    }
    run_gemm(stream, fB, PF(iP2), PF(iP2 + 1), C1, BB, 256, 256, 512, 512, 1, wbuf);
    {
        long n = (long)BB * 256;
        k_sig_f16<<<cdiv(n, 256), 256, 0, stream>>>(C1, 256, 256, fA, 256, n);
    }
    run_gemm(stream, fA, PF(iP3), PF(iP3 + 1), C0, BB, 1, 16, 256, 256, 1, wbuf);
    k_final_out<<<cdiv(BB, 256), 256, 0, stream>>>(C0, out, BB);

    // ---- reverse loss ----
    k_rev_loss<<<1, 1024, 0, stream>>>(UfRev, UdRev, sensitive, out + BB, BB);
#undef PF
}
